// UniModelMatcher_83434034692461
// MI455X (gfx1250) — matrix operand order repaired and checked by reading, not compiled, not run on hardware
//
#include <hip/hip_runtime.h>
#include <hip/hip_bf16.h>

// ---------------------------------------------------------------------------
// CDNA5 (gfx1250) wave32 WMMA implementation of the entity-mention matcher.
// ---------------------------------------------------------------------------

typedef __attribute__((ext_vector_type(16))) __bf16 v16bf;
typedef __attribute__((ext_vector_type(8)))  float  v8f;

union FragB16 { v16bf v; unsigned int u[8]; };

__device__ __forceinline__ unsigned short f2bf(float f) {
  unsigned int x = __float_as_uint(f);
  x += 0x7fffu + ((x >> 16) & 1u);          // round-to-nearest-even
  return (unsigned short)(x >> 16);
}

// A-matrix fragment (16x32 bf16, row-major source, this lane's row pointer p).
// ISA layout: lanes 0-15 hold K = {0..7, 16..23}; lanes 16-31 K = {8..15, 24..31}.
__device__ __forceinline__ void load_fragA(FragB16& f, const unsigned short* p, int half) {
#pragma unroll
  for (int j = 0; j < 8; ++j) {
    int kk = ((j & 4) << 2) + (half << 3) + ((j & 3) << 1); // (j<4?0:16)+8*half+2*(j&3)
    f.u[j] = *(const unsigned int*)(p + kk);
  }
}

// B-matrix fragment (32x16 bf16). Source stored k-contiguous per column (Bt[n][k]).
// ISA layout: lanes 0-15 hold K = {0..7, 16..23}; lanes 16-31 K = {8..15, 24..31}.
__device__ __forceinline__ void load_fragB(FragB16& f, const unsigned short* p, int half) {
#pragma unroll
  for (int j = 0; j < 8; ++j) {
    int kk = ((j & 4) << 2) + (half << 3) + ((j & 3) << 1);
    f.u[j] = *(const unsigned int*)(p + kk);
  }
}

__device__ __forceinline__ v8f wmma_bf16(const FragB16& a, const FragB16& b, v8f c) {
  return __builtin_amdgcn_wmma_f32_16x16x32_bf16(false, a.v, false, b.v, (short)0, c,
                                                 false, false);
}

__device__ __forceinline__ float wave_sum(float v) {
#pragma unroll
  for (int m = 16; m >= 1; m >>= 1) v += __shfl_xor(v, m, 32);
  return v;
}

// ---------------------------------------------------------------------------
// Conversion kernels
// ---------------------------------------------------------------------------
__global__ void cvt_bf16(const float* __restrict__ X, unsigned short* __restrict__ Y, int n) {
  int i = blockIdx.x * 256 + threadIdx.x;
  if (i < n) Y[i] = f2bf(X[i]);
}

// Wt[n*768 + k] = W[k*768 + n]   (weights -> [N,K] bf16 for B-fragments)
__global__ void cvt_transpose768(const float* __restrict__ W, unsigned short* __restrict__ Wt) {
  int t = blockIdx.x * 256 + threadIdx.x;     // 768*768 elements
  int n = t / 768, k = t - n * 768;
  Wt[t] = f2bf(W[k * 768 + n]);
}

// ---------------------------------------------------------------------------
// Generic GEMM:  Cout[M,N] = A[M,K](bf16) * Bt[N,K](bf16)^T + bias[N]
// Optionally also emits bf16 k-transposed copy Vt[(m/32), n, m%32] (for T@v GEMM).
// One wave computes one 16x16 tile; 4 waves per block.
// ---------------------------------------------------------------------------
__global__ void gemm_bias_bf16(const unsigned short* __restrict__ A,
                               const unsigned short* __restrict__ Bt,
                               const float* __restrict__ bias,
                               float* __restrict__ Cout,
                               unsigned short* __restrict__ Vt,
                               int ntiles, int ntn, int N, int K) {
  int tile = blockIdx.x * (blockDim.x >> 5) + (threadIdx.x >> 5);
  if (tile >= ntiles) return;
  int tm = tile / ntn, tn = tile - tm * ntn;
  int lane = threadIdx.x & 31;
  int half = lane >> 4, idx = lane & 15;
  const unsigned short* ap = A  + (tm * 16 + idx) * K;   // this lane's A row
  const unsigned short* bp = Bt + (tn * 16 + idx) * K;   // this lane's output column
  v8f acc = {};
  for (int k0 = 0; k0 < K; k0 += 32) {
    __builtin_prefetch(ap + k0 + 128, 0, 3);   // locality 3 -> WGP-scope prefetch
    __builtin_prefetch(bp + k0 + 128, 0, 3);
    FragB16 a, b;
    load_fragA(a, ap + k0, half);
    load_fragB(b, bp + k0, half);
    acc = wmma_bf16(a, b, acc);
  }
  int n = tn * 16 + idx;
  float bb = bias ? bias[n] : 0.0f;
#pragma unroll
  for (int r = 0; r < 8; ++r) {
    int m = tm * 16 + r + half * 8;
    float val = acc[r] + bb;
    Cout[m * N + n] = val;
    if (Vt) Vt[(((m >> 5) * N + n) << 5) + (m & 31)] = f2bf(val);
  }
}

// ---------------------------------------------------------------------------
// Row L2 normalize (fp32 [rows,768] -> bf16), one wave per row.
// ---------------------------------------------------------------------------
__global__ void rownorm_bf16(const float* __restrict__ X, unsigned short* __restrict__ Y) {
  int row  = blockIdx.x * 8 + (threadIdx.x >> 5);
  int lane = threadIdx.x & 31;
  const float* xr = X + row * 768;
  float ss = 0.0f;
  for (int c = lane; c < 768; c += 32) { float v = xr[c]; ss += v * v; }
  ss = wave_sum(ss);
  float inv = 1.0f / fmaxf(sqrtf(ss), 1e-8f);
  for (int c = lane; c < 768; c += 32) Y[row * 768 + c] = f2bf(xr[c] * inv);
}

// ---------------------------------------------------------------------------
// Cost GEMM: C[a,b,i,j] = qn[a*32+i,:] . kn[b*32+j,:]    (K = 768)
// ---------------------------------------------------------------------------
__global__ void gemm_cost(const unsigned short* __restrict__ Qn,   // [1024,768]
                          const unsigned short* __restrict__ Kn,   // [2048,768]
                          float* __restrict__ C) {                 // [32,64,32,32]
  int tile = blockIdx.x * (blockDim.x >> 5) + (threadIdx.x >> 5);
  if (tile >= 64 * 128) return;
  int tm = tile >> 7, tn = tile & 127;
  int lane = threadIdx.x & 31;
  int half = lane >> 4, idx = lane & 15;
  const unsigned short* ap = Qn + (tm * 16 + idx) * 768;
  const unsigned short* bp = Kn + (tn * 16 + idx) * 768;
  v8f acc = {};
  for (int k0 = 0; k0 < 768; k0 += 32) {
    __builtin_prefetch(ap + k0 + 128, 0, 3);
    __builtin_prefetch(bp + k0 + 128, 0, 3);
    FragB16 a, b;
    load_fragA(a, ap + k0, half);
    load_fragB(b, bp + k0, half);
    acc = wmma_bf16(a, b, acc);
  }
  int n = tn * 16 + idx;
  int bb = n >> 5, j = n & 31;
#pragma unroll
  for (int r = 0; r < 8; ++r) {
    int m = tm * 16 + r + half * 8;
    int a = m >> 5, i = m & 31;
    C[((((a << 6) + bb) << 10)) + (i << 5) + j] = acc[r];
  }
}

// ---------------------------------------------------------------------------
// Sinkhorn: one block per (a,b), 1024 threads = 32 wave32 = one wave per row.
// ---------------------------------------------------------------------------
__global__ void sinkhorn_kernel(const float* __restrict__ C, unsigned short* __restrict__ Tbf) {
  __shared__ float tile[32 * 33];
  int ab = blockIdx.x;
  int i = threadIdx.x >> 5, j = threadIdx.x & 31;
  float q = __expf(C[(ab << 10) + (i << 5) + j] * 10.0f);   // exp(C / 0.1)
  for (int it = 0; it < 10; ++it) {
    float rs = wave_sum(q);                 // row (last-axis) sum, whole wave = one row
    q /= rs;
    tile[i * 33 + j] = q;
    __syncthreads();
    float cs = 0.0f;
#pragma unroll
    for (int ii = 0; ii < 32; ++ii) cs += tile[ii * 33 + j];   // column sum
    q /= cs;
    __syncthreads();
  }
  float ss = wave_sum(q * q);
  q /= fmaxf(sqrtf(ss), 1e-12f);            // L2 normalize last dim
  Tbf[(ab << 10) + (i << 5) + j] = f2bf(q);
}

// ---------------------------------------------------------------------------
// Fused attention + LN1 + softpool + LN2 -> context.  One block per (a,b),
// 256 threads (8 waves). attended tile (32x768 fp32 = 96KB) lives in LDS.
// ---------------------------------------------------------------------------
__global__ void attend_ctx(const unsigned short* __restrict__ Tbf,  // [2048,32,32]
                           const unsigned short* __restrict__ Vt,   // [64,768,32]
                           const float* __restrict__ Qf,            // [1024,768]
                           const float* __restrict__ ln1g, const float* __restrict__ ln1b,
                           const float* __restrict__ Wsp,  const float* __restrict__ bsp,
                           const float* __restrict__ ln2g, const float* __restrict__ ln2b,
                           float* __restrict__ Ctx) {               // [2048,768]
  extern __shared__ float smem[];
  float* att  = smem;                 // 32*768
  float* srow = smem + 32 * 768;      // 32
  float* wrow = srow + 32;            // 32
  float* reds = wrow + 32;            // 8
  float* redq = reds + 8;             // 8

  int ab = blockIdx.x;
  int a = ab >> 6, b = ab & 63;
  int tid = threadIdx.x, wid = tid >> 5, lane = tid & 31;
  int half = lane >> 4, idx = lane & 15;

  // ---- attended = T @ v + q  (K = 32, single WMMA per 16x16 tile) ----
  int mhalf = wid & 1;
  FragB16 afr;
  load_fragA(afr, Tbf + (ab << 10) + (mhalf * 16 + idx) * 32, half);
  const unsigned short* vbase = Vt + ((b * 768) << 5);
  for (int nt = (wid >> 1); nt < 48; nt += 4) {
    FragB16 bfr;
    load_fragB(bfr, vbase + ((nt * 16 + idx) << 5), half);
    v8f acc = {};
    acc = wmma_bf16(afr, bfr, acc);
    int n = nt * 16 + idx;
#pragma unroll
    for (int r = 0; r < 8; ++r) {
      int m = mhalf * 16 + r + half * 8;
      att[m * 768 + n] = acc[r] + Qf[(a * 32 + m) * 768 + n];
    }
  }
  __syncthreads();

  // ---- layernorm1 per row + softpool score; 8 waves x 4 rows ----
  for (int row = wid * 4; row < wid * 4 + 4; ++row) {
    float s = 0.0f, q2 = 0.0f;
    for (int c = lane; c < 768; c += 32) { float v = att[row * 768 + c]; s += v; q2 += v * v; }
    s = wave_sum(s); q2 = wave_sum(q2);
    float mean = s * (1.0f / 768.0f);
    float var  = q2 * (1.0f / 768.0f) - mean * mean;
    float inv  = rsqrtf(var + 1e-5f);
    float dot = 0.0f;
    for (int c = lane; c < 768; c += 32) {
      float v = (att[row * 768 + c] - mean) * inv * ln1g[c] + ln1b[c];
      att[row * 768 + c] = v;
      dot += v * Wsp[c];
    }
    dot = wave_sum(dot);
    if (lane == 0) srow[row] = dot + bsp[0];
  }
  __syncthreads();

  // ---- softmax over token dim ----
  if (tid < 32) {
    float mx = -1e30f;
    for (int r = 0; r < 32; ++r) mx = fmaxf(mx, srow[r]);
    float se = 0.0f;
    for (int r = 0; r < 32; ++r) se += __expf(srow[r] - mx);
    wrow[tid] = __expf(srow[tid] - mx) / se;
  }
  __syncthreads();

  // ---- pooled = sum_i att[i,:] * w[i]; then layernorm2 -> context ----
  float p[3];
  float ls = 0.0f, lq = 0.0f;
#pragma unroll
  for (int t = 0; t < 3; ++t) {
    int n = tid + t * 256;
    float acc = 0.0f;
#pragma unroll
    for (int r = 0; r < 32; ++r) acc += att[r * 768 + n] * wrow[r];
    p[t] = acc; ls += acc; lq += acc * acc;
  }
  ls = wave_sum(ls); lq = wave_sum(lq);
  if (lane == 0) { reds[wid] = ls; redq[wid] = lq; }
  __syncthreads();
  float ts = 0.0f, tq = 0.0f;
#pragma unroll
  for (int w = 0; w < 8; ++w) { ts += reds[w]; tq += redq[w]; }
  float mean = ts * (1.0f / 768.0f);
  float var  = tq * (1.0f / 768.0f) - mean * mean;
  float inv  = rsqrtf(var + 1e-5f);
#pragma unroll
  for (int t = 0; t < 3; ++t) {
    int n = tid + t * 256;
    Ctx[ab * 768 + n] = (p[t] - mean) * inv * ln2g[n] + ln2b[n];
  }
}

// ---------------------------------------------------------------------------
// Final scores: out[b,a] = 0.5*(ecls[a].ctx[a,b] + mention_cls[b].entity_cls[a])
// One wave per output element.
// ---------------------------------------------------------------------------
__global__ void final_scores(const float* __restrict__ Ecls,   // [32,768]
                             const float* __restrict__ Ctx,    // [32,64,768]
                             const float* __restrict__ EntCls, // [32,768]
                             const float* __restrict__ MenCls, // [64,768]
                             float* __restrict__ Out) {        // [64,32]
  int g = blockIdx.x * 8 + (threadIdx.x >> 5);
  int lane = threadIdx.x & 31;
  int b = g >> 5, a = g & 31;
  const float* e  = Ecls   + a * 768;
  const float* c  = Ctx    + (a * 64 + b) * 768;
  const float* ec = EntCls + a * 768;
  const float* mc = MenCls + b * 768;
  float d = 0.0f;
  for (int h = lane; h < 768; h += 32) d += e[h] * c[h] + ec[h] * mc[h];
  d = wave_sum(d);
  if (lane == 0) Out[b * 32 + a] = 0.5f * d;
}

// ---------------------------------------------------------------------------
// Host launcher
// ---------------------------------------------------------------------------
extern "C" void kernel_launch(void* const* d_in, const int* in_sizes, int n_in,
                              void* d_out, int out_size, void* d_ws, size_t ws_size,
                              hipStream_t stream) {
  (void)in_sizes; (void)n_in; (void)out_size; (void)ws_size;

  const float* entity_cls     = (const float*)d_in[0];   // [32,768]
  const float* entity_tokens  = (const float*)d_in[1];   // [32,32,768]
  const float* mention_cls    = (const float*)d_in[2];   // [64,768]
  const float* mention_tokens = (const float*)d_in[3];   // [64,32,768]
  const float* Wq   = (const float*)d_in[4];
  const float* bq   = (const float*)d_in[5];
  const float* Wk   = (const float*)d_in[6];
  const float* bk   = (const float*)d_in[7];
  const float* Wv   = (const float*)d_in[8];
  const float* bv   = (const float*)d_in[9];
  const float* ln1g = (const float*)d_in[10];
  const float* ln1b = (const float*)d_in[11];
  const float* Wcls = (const float*)d_in[12];
  const float* bcls = (const float*)d_in[13];
  const float* Wsp  = (const float*)d_in[14];
  const float* bsp  = (const float*)d_in[15];
  const float* ln2g = (const float*)d_in[16];
  const float* ln2b = (const float*)d_in[17];
  float* Out = (float*)d_out;

  // ---- workspace carve-out (all regions multiple of 256B) ----
  char* ws = (char*)d_ws;
  size_t off = 0;
  auto alloc = [&](size_t bytes) { char* p = ws + off; off += (bytes + 255) & ~(size_t)255; return p; };
  unsigned short* Et   = (unsigned short*)alloc(1024u * 768u * 2u);   // entity_tokens bf16
  unsigned short* Mt   = (unsigned short*)alloc(2048u * 768u * 2u);   // mention_tokens bf16
  unsigned short* Wqt  = (unsigned short*)alloc(768u * 768u * 2u);    // Wq^T bf16
  unsigned short* Wkt  = (unsigned short*)alloc(768u * 768u * 2u);
  unsigned short* Wvt  = (unsigned short*)alloc(768u * 768u * 2u);
  unsigned short* Wct  = (unsigned short*)alloc(768u * 768u * 2u);
  unsigned short* Ecb  = (unsigned short*)alloc(32u * 768u * 2u);     // entity_cls bf16
  float*          Qf   = (float*)alloc(1024u * 768u * 4u);            // q  fp32
  float*          Kf   = (float*)alloc(2048u * 768u * 4u);            // k  fp32
  float*          Vf   = (float*)alloc(2048u * 768u * 4u);            // v  fp32
  float*          Ef   = (float*)alloc(32u * 768u * 4u);              // ecls fp32
  unsigned short* Qn   = (unsigned short*)alloc(1024u * 768u * 2u);   // qn bf16
  unsigned short* Kn   = (unsigned short*)alloc(2048u * 768u * 2u);   // kn bf16
  unsigned short* Vt   = (unsigned short*)alloc(2048u * 768u * 2u);   // v  bf16, [b][n][k]
  float*          Cc   = (float*)alloc(2048u * 1024u * 4u);           // cost [a,b,i,j]
  unsigned short* Tb   = (unsigned short*)alloc(2048u * 1024u * 2u);  // transport bf16
  float*          Ctx  = (float*)alloc(2048u * 768u * 4u);            // context fp32

  // 1) conversions
  cvt_bf16<<<3072, 256, 0, stream>>>(entity_tokens, Et, 1024 * 768);
  cvt_bf16<<<6144, 256, 0, stream>>>(mention_tokens, Mt, 2048 * 768);
  cvt_bf16<<<96,   256, 0, stream>>>(entity_cls, Ecb, 32 * 768);
  cvt_transpose768<<<2304, 256, 0, stream>>>(Wq,   Wqt);
  cvt_transpose768<<<2304, 256, 0, stream>>>(Wk,   Wkt);
  cvt_transpose768<<<2304, 256, 0, stream>>>(Wv,   Wvt);
  cvt_transpose768<<<2304, 256, 0, stream>>>(Wcls, Wct);

  // 2) projection GEMMs (WMMA bf16, fp32 accumulate)
  gemm_bias_bf16<<<768,  128, 0, stream>>>(Et,  Wqt, bq,   Qf, nullptr, 64 * 48,  48, 768, 768);
  gemm_bias_bf16<<<1536, 128, 0, stream>>>(Mt,  Wkt, bk,   Kf, nullptr, 128 * 48, 48, 768, 768);
  gemm_bias_bf16<<<1536, 128, 0, stream>>>(Mt,  Wvt, bv,   Vf, Vt,      128 * 48, 48, 768, 768);
  gemm_bias_bf16<<<24,   128, 0, stream>>>(Ecb, Wct, bcls, Ef, nullptr, 2 * 48,   48, 768, 768);

  // 3) cosine normalization + cost GEMM
  rownorm_bf16<<<128, 256, 0, stream>>>(Qf, Qn);
  rownorm_bf16<<<256, 256, 0, stream>>>(Kf, Kn);
  gemm_cost<<<2048, 128, 0, stream>>>(Qn, Kn, Cc);

  // 4) Sinkhorn transport
  sinkhorn_kernel<<<2048, 1024, 0, stream>>>(Cc, Tb);

  // 5) fused attention + LN1 + softpool + LN2
  size_t shmem = (size_t)(32 * 768 + 32 + 32 + 8 + 8) * sizeof(float);
  attend_ctx<<<2048, 256, shmem, stream>>>(Tb, Vt, Qf, ln1g, ln1b, Wsp, bsp, ln2g, ln2b, Ctx);

  // 6) final matching scores
  final_scores<<<256, 256, 0, stream>>>(Ef, Ctx, entity_cls, mention_cls, Out);
}
